// DeepseekV3Attention_80779744903837
// MI455X (gfx1250) — matrix operand order repaired and checked by reading, not compiled, not run on hardware
//
#include <hip/hip_runtime.h>
#include <hip/hip_bf16.h>

// ---------------------------------------------------------------------------
// DeepSeek-V3 MLA attention for gfx1250 (MI455X).
// bf16 WMMA pipeline + async global->LDS staging (double buffered).
// ---------------------------------------------------------------------------

typedef __attribute__((ext_vector_type(16))) __bf16 v16bf;
typedef __attribute__((ext_vector_type(8)))  __bf16 bf16x8;
typedef __attribute__((ext_vector_type(4)))  __bf16 bf16x4;
typedef __attribute__((ext_vector_type(8)))  float  v8f;
typedef __attribute__((ext_vector_type(4)))  int    v4i;

namespace cfg {
constexpr int kB   = 2;
constexpr int kS   = 2048;
constexpr int kNH  = 16;
constexpr int kDN  = 128;
constexpr int kDR  = 64;
constexpr int kDV  = 128;
constexpr float kScale = 0.07216878364870323f;  // 1/sqrt(192)
}

// ------------------------- async copy (CDNA5) ------------------------------

#if __has_builtin(__builtin_amdgcn_global_load_async_to_lds_b128)
#define HAVE_ASYNC 1
#else
#define HAVE_ASYNC 0
#endif

static __device__ __forceinline__ void async_b128(void* dst_lds, const void* src_g) {
#if HAVE_ASYNC
  typedef __attribute__((address_space(1))) v4i GV;
  typedef __attribute__((address_space(3))) v4i LV;
  __builtin_amdgcn_global_load_async_to_lds_b128(
      (GV*)(v4i*)const_cast<void*>(src_g),
      (LV*)(v4i*)dst_lds, 0, 0);
#else
  *(bf16x8*)dst_lds = *(const bf16x8*)src_g;
#endif
}

static __device__ __forceinline__ void async_wait() {
#if HAVE_ASYNC
#if __has_builtin(__builtin_amdgcn_s_wait_asynccnt)
  __builtin_amdgcn_s_wait_asynccnt(0);
#else
  asm volatile("s_wait_asynccnt 0" ::: "memory");
#endif
#endif
}

// ------------------------- small helpers -----------------------------------

static __device__ __forceinline__ __bf16 f2bf(float f) { return (__bf16)f; }
static __device__ __forceinline__ float  bf2f(__bf16 b) { return (float)b; }

static __device__ __forceinline__ v16bf pack16(bf16x8 lo, bf16x8 hi) {
  v16bf r;
#pragma unroll
  for (int i = 0; i < 8; ++i) { r[i] = lo[i]; r[i + 8] = hi[i]; }
  return r;
}

// ---------------------------------------------------------------------------
// Pre-pack kernels: f32 -> bf16 (identity layout) and f32 -> bf16 transpose.
// ---------------------------------------------------------------------------

__global__ __launch_bounds__(256) void cvt_bf16(const float* __restrict__ in,
                                                void* __restrict__ out_, int n4) {
  const int i = blockIdx.x * blockDim.x + threadIdx.x;
  if (i >= n4) return;
  const float4 f = ((const float4*)in)[i];
  bf16x4 r;
  r[0] = f2bf(f.x); r[1] = f2bf(f.y); r[2] = f2bf(f.z); r[3] = f2bf(f.w);
  ((bf16x4*)out_)[i] = r;
}

// in: [K][N] f32 row-major  ->  out: [N][K] bf16 row-major
__global__ __launch_bounds__(256) void transpose_w(const float* __restrict__ in,
                                                   void* __restrict__ out_,
                                                   int K, int N) {
  __shared__ float tile[32][33];
  const int n0 = blockIdx.x * 32;
  const int k0 = blockIdx.y * 32;
  const int tx = threadIdx.x & 31;
  const int ty = threadIdx.x >> 5;   // 0..7
  __bf16* out = (__bf16*)out_;
#pragma unroll
  for (int j = 0; j < 4; ++j)
    tile[ty + j * 8][tx] = in[(size_t)(k0 + ty + j * 8) * N + n0 + tx];
  __syncthreads();
#pragma unroll
  for (int j = 0; j < 4; ++j)
    out[(size_t)(n0 + ty + j * 8) * K + k0 + tx] = f2bf(tile[tx][ty + j * 8]);
}

// vbuf [B*S][NH*DV] bf16 -> vt [B][NH][DV][S] bf16  (per-head transpose)
__global__ __launch_bounds__(256) void transpose_v(const void* __restrict__ in_,
                                                   void* __restrict__ out_) {
  using namespace cfg;
  __shared__ __bf16 tile[32][40];
  const __bf16* in = (const __bf16*)in_;
  __bf16* out = (__bf16*)out_;
  const int bh = blockIdx.x;             // 0..31
  const int b = bh >> 4, h = bh & 15;
  const int s0 = blockIdx.y * 32;
  const int d0 = blockIdx.z * 32;
  const int tx = threadIdx.x & 31;
  const int ty = threadIdx.x >> 5;
#pragma unroll
  for (int j = 0; j < 4; ++j)
    tile[ty + j * 8][tx] =
        in[((size_t)b * kS + s0 + ty + j * 8) * (kNH * kDV) + h * kDV + d0 + tx];
  __syncthreads();
#pragma unroll
  for (int j = 0; j < 4; ++j)
    out[(((size_t)b * kNH + h) * kDV + d0 + ty + j * 8) * kS + s0 + tx] =
        tile[tx][ty + j * 8];
}

// ---------------------------------------------------------------------------
// NT GEMM: C[M,N] = A[M,K] * Bt[N,K]^T, both operands bf16 with K contiguous.
// BM=BN=128, BK=32, 8 waves; each wave owns 32x64 C (2x4 WMMA tiles).
// Double-buffered LDS fed by async global->LDS b128 copies.
// CMODE: 0 = f32 C, 1 = bf16 C.
// ---------------------------------------------------------------------------

#define G_LDS 40   // padded stride (bf16): 80 B = 5*16 B

template <int CMODE>
__global__ __launch_bounds__(256) void gemm_nt(const void* __restrict__ A_,
                                               const void* __restrict__ Bt_,
                                               void* __restrict__ C_,
                                               int N, int K) {
  __shared__ __bf16 As[2][128 * G_LDS];
  __shared__ __bf16 Bs[2][128 * G_LDS];

  const __bf16* A  = (const __bf16*)A_;
  const __bf16* Bt = (const __bf16*)Bt_;
  const int tid  = threadIdx.x;
  const int wave = tid >> 5;
  const int lane = tid & 31;
  const int l16  = lane & 15;
  const int lh   = lane >> 4;
  const int wm   = wave >> 1;            // 0..3
  const int wn   = wave & 1;             // 0..1
  const size_t bm = (size_t)blockIdx.x * 128;
  const size_t bn = (size_t)blockIdx.y * 128;

  v8f acc[2][4] = {};

  auto stage = [&](int buf, int k0) {
#pragma unroll
    for (int j = 0; j < 2; ++j) {
      const int u = tid * 2 + j;         // 0..511
      const int row = u >> 2;            // 0..127
      const int seg = (u & 3) * 8;       // 0,8,16,24 (bf16 elems)
      async_b128(&As[buf][row * G_LDS + seg], A + (bm + row) * (size_t)K + k0 + seg);
      async_b128(&Bs[buf][row * G_LDS + seg], Bt + (bn + row) * (size_t)K + k0 + seg);
    }
  };

  const int nk = K >> 5;
  stage(0, 0);
  async_wait();
  __syncthreads();

  for (int kc = 0; kc < nk; ++kc) {
    const int cur = kc & 1;
    if (kc + 1 < nk) stage(cur ^ 1, (kc + 1) * 32);

    v16bf af[2], bfv[4];
#pragma unroll
    for (int m = 0; m < 2; ++m) {
      const __bf16* p = &As[cur][(wm * 32 + m * 16 + l16) * G_LDS + lh * 8];
      af[m] = pack16(*(const bf16x8*)p, *(const bf16x8*)(p + 16));
    }
#pragma unroll
    for (int n = 0; n < 4; ++n) {
      const __bf16* p = &Bs[cur][(wn * 64 + n * 16 + l16) * G_LDS + lh * 8];
      bfv[n] = pack16(*(const bf16x8*)p, *(const bf16x8*)(p + 16));
    }
#pragma unroll
    for (int m = 0; m < 2; ++m)
#pragma unroll
      for (int n = 0; n < 4; ++n)
        acc[m][n] = __builtin_amdgcn_wmma_f32_16x16x32_bf16(
            false, af[m], false, bfv[n], (short)0, acc[m][n], false, false);

    async_wait();
    __syncthreads();
  }

#pragma unroll
  for (int m = 0; m < 2; ++m)
#pragma unroll
    for (int n = 0; n < 4; ++n)
#pragma unroll
      for (int i = 0; i < 8; ++i) {
        const size_t row = bm + wm * 32 + m * 16 + i + lh * 8;
        const size_t col = bn + wn * 64 + n * 16 + l16;
        const float val = acc[m][n][i];
        if constexpr (CMODE == 1) ((__bf16*)C_)[row * N + col] = f2bf(val);
        else                      ((float*)C_)[row * N + col]  = val;
      }
}

// ---------------------------------------------------------------------------
// RoPE, in place on bf16 [rows][NH*DR]; one thread per (d, d+32) pair.
// ---------------------------------------------------------------------------

__global__ __launch_bounds__(256) void rope_kernel(void* x_, int total_pairs) {
  const int idx = blockIdx.x * blockDim.x + threadIdx.x;
  if (idx >= total_pairs) return;
  __bf16* x = (__bf16*)x_;
  const int p   = idx & 31;
  const int h   = (idx >> 5) & (cfg::kNH - 1);
  const int row = idx >> 9;
  const int pos = row & (cfg::kS - 1);
  const float inv = __powf(10000.f, -(float)p * (1.0f / 32.0f));
  float sn, cs;
  __sincosf((float)pos * inv, &sn, &cs);
  __bf16* base = x + (size_t)row * (cfg::kNH * cfg::kDR) + h * cfg::kDR;
  const float x0 = bf2f(base[p]);
  const float x1 = bf2f(base[p + 32]);
  base[p]      = f2bf(x0 * cs - x1 * sn);
  base[p + 32] = f2bf(x1 * cs + x0 * sn);
}

// ---------------------------------------------------------------------------
// Causal flash attention.  grid = (B*NH, S/128); 8 waves x 16 query rows.
// K_nope / K_rope / V^T staged per 32-key chunk via async b128 copies into
// double-buffered LDS.  Q fragments in registers; P re-striped via a
// per-wave LDS round-trip; 20 WMMAs per chunk per wave.
// ---------------------------------------------------------------------------

__global__ __launch_bounds__(256) void mla_attn(const void* __restrict__ qn_,
                                                const void* __restrict__ qr_,
                                                const void* __restrict__ kn_,
                                                const void* __restrict__ kr_,
                                                const void* __restrict__ vt_,
                                                void* __restrict__ ctx_) {
  using namespace cfg;
  const __bf16* qn = (const __bf16*)qn_;
  const __bf16* qr = (const __bf16*)qr_;
  const __bf16* kn = (const __bf16*)kn_;
  const __bf16* kr = (const __bf16*)kr_;
  const __bf16* vt = (const __bf16*)vt_;

  __shared__ __bf16 sKn[2][32 * 136];   // [key][dim], 272 B stride
  __shared__ __bf16 sKr[2][32 * 72];    // [key][dim], 144 B stride
  __shared__ __bf16 sVt[2][128 * 40];   // [dv][key],   80 B stride
  __shared__ __bf16 sP[8 * 512];        // per-wave 16x32 P staging

  const int bh = blockIdx.x;
  const int b  = bh >> 4;
  const int h  = bh & 15;
  const int q0 = blockIdx.y * 128;
  const int tid  = threadIdx.x;
  const int wave = tid >> 5;
  const int lane = tid & 31;
  const int l16  = lane & 15;
  const int lh   = lane >> 4;

  // ---- Q fragments (A-matrix layout) ----
  v16bf aq[6];
  {
    const size_t qrow = (size_t)b * kS + q0 + wave * 16 + l16;
    const __bf16* qnp = qn + qrow * (kNH * kDN) + h * kDN;
    const __bf16* qrp = qr + qrow * (kNH * kDR) + h * kDR;
#pragma unroll
    for (int c = 0; c < 4; ++c) {
      const __bf16* p = qnp + c * 32 + lh * 8;
      aq[c] = pack16(*(const bf16x8*)p, *(const bf16x8*)(p + 16));
    }
#pragma unroll
    for (int c = 0; c < 2; ++c) {
      const __bf16* p = qrp + c * 32 + lh * 8;
      aq[4 + c] = pack16(*(const bf16x8*)p, *(const bf16x8*)(p + 16));
    }
  }

  v8f acc[8] = {};
  float mi[8], li[8];
#pragma unroll
  for (int i = 0; i < 8; ++i) { mi[i] = -3.0e38f; li[i] = 0.0f; }

  auto stage = [&](int buf, int kt) {
    const size_t krow = (size_t)b * kS + kt;
    const size_t vtbase = (((size_t)b * kNH + h) * kDV) * kS + kt;
#pragma unroll
    for (int j = 0; j < 2; ++j) {
      const int u = tid * 2 + j;                    // 0..511
      { const int key = u >> 4, seg = (u & 15) * 8; // 32 x 128 dims
        async_b128(&sKn[buf][key * 136 + seg],
                   kn + (krow + key) * (kNH * kDN) + h * kDN + seg); }
      { const int dv = u >> 2, seg = (u & 3) * 8;   // 128 x 32 keys
        async_b128(&sVt[buf][dv * 40 + seg],
                   vt + vtbase + (size_t)dv * kS + seg); }
    }
    { const int key = tid >> 3, seg = (tid & 7) * 8; // 32 x 64 dims
      async_b128(&sKr[buf][key * 72 + seg],
                 kr + (krow + key) * (kNH * kDR) + h * kDR + seg); }
  };

  const int nchunks = (q0 >> 5) + 4;
  stage(0, 0);
  async_wait();
  __syncthreads();

  for (int kc = 0; kc < nchunks; ++kc) {
    const int cur = kc & 1;
    const int kt = kc * 32;
    if (kc + 1 < nchunks) stage(cur ^ 1, (kc + 1) * 32);

    // ---- scores: two 16x16 tiles, 6 WMMAs each ----
    v8f s[2] = {};
#pragma unroll
    for (int t = 0; t < 2; ++t) {
      const int key = t * 16 + l16;
#pragma unroll
      for (int c = 0; c < 4; ++c) {
        const __bf16* p = &sKn[cur][key * 136 + c * 32 + lh * 8];
        v16bf bfr = pack16(*(const bf16x8*)p, *(const bf16x8*)(p + 16));
        s[t] = __builtin_amdgcn_wmma_f32_16x16x32_bf16(
            false, aq[c], false, bfr, (short)0, s[t], false, false);
      }
#pragma unroll
      for (int c = 0; c < 2; ++c) {
        const __bf16* p = &sKr[cur][key * 72 + c * 32 + lh * 8];
        v16bf bfr = pack16(*(const bf16x8*)p, *(const bf16x8*)(p + 16));
        s[t] = __builtin_amdgcn_wmma_f32_16x16x32_bf16(
            false, aq[4 + c], false, bfr, (short)0, s[t], false, false);
      }
    }

    // ---- scale + causal mask + online softmax ----
#pragma unroll
    for (int i = 0; i < 8; ++i) {
      const int qq = q0 + wave * 16 + i + lh * 8;
      const int kk = kt + l16;
      float s0 = s[0][i] * kScale + ((kk <= qq) ? 0.0f : -10000.0f);
      float s1 = s[1][i] * kScale + ((kk + 16 <= qq) ? 0.0f : -10000.0f);
      float mx = fmaxf(s0, s1);
#pragma unroll
      for (int off = 8; off > 0; off >>= 1) mx = fmaxf(mx, __shfl_xor(mx, off, 16));
      const float mnew = fmaxf(mi[i], mx);
      const float corr = __expf(mi[i] - mnew);
      const float p0 = __expf(s0 - mnew);
      const float p1 = __expf(s1 - mnew);
      float rsum = p0 + p1;
#pragma unroll
      for (int off = 8; off > 0; off >>= 1) rsum += __shfl_xor(rsum, off, 16);
      li[i] = li[i] * corr + rsum;
      mi[i] = mnew;
#pragma unroll
      for (int n = 0; n < 8; ++n) acc[n][i] *= corr;
      s[0][i] = p0;
      s[1][i] = p1;
    }

    // ---- P: C-layout -> bf16 A-layout via per-wave LDS round-trip ----
    __bf16* Pw = &sP[wave * 512];
#pragma unroll
    for (int i = 0; i < 8; ++i) {
      Pw[(i + lh * 8) * 32 + l16]      = f2bf(s[0][i]);
      Pw[(i + lh * 8) * 32 + 16 + l16] = f2bf(s[1][i]);
    }
    __builtin_amdgcn_wave_barrier();
    v16bf pf;
    {
      const __bf16* p = &Pw[l16 * 32 + lh * 8];
      pf = pack16(*(const bf16x8*)p, *(const bf16x8*)(p + 16));
    }

    // ---- out += P(16x32) @ V(32x128): 8 WMMAs ----
#pragma unroll
    for (int n = 0; n < 8; ++n) {
      const __bf16* p = &sVt[cur][(n * 16 + l16) * 40 + lh * 8];
      v16bf bfr = pack16(*(const bf16x8*)p, *(const bf16x8*)(p + 16));
      acc[n] = __builtin_amdgcn_wmma_f32_16x16x32_bf16(
          false, pf, false, bfr, (short)0, acc[n], false, false);
    }

    async_wait();
    __syncthreads();
  }

  // ---- epilogue ----
  {
    const size_t qbase = (size_t)b * kS + q0 + wave * 16;
    __bf16* ctx = (__bf16*)ctx_;
#pragma unroll
    for (int n = 0; n < 8; ++n)
#pragma unroll
      for (int i = 0; i < 8; ++i) {
        const size_t r = qbase + i + lh * 8;
        ctx[r * (kNH * kDV) + h * kDV + n * 16 + l16] = f2bf(acc[n][i] / li[i]);
      }
  }
}

// ---------------------------------------------------------------------------
// Host launcher
// ---------------------------------------------------------------------------

extern "C" void kernel_launch(void* const* d_in, const int* in_sizes, int n_in,
                              void* d_out, int out_size, void* d_ws, size_t ws_size,
                              hipStream_t stream) {
  (void)in_sizes; (void)n_in; (void)out_size; (void)ws_size;
  using namespace cfg;

  const float* hidden   = (const float*)d_in[0];
  const float* w_q_a    = (const float*)d_in[2];
  const float* w_q_nope = (const float*)d_in[3];
  const float* w_q_rope = (const float*)d_in[4];
  const float* w_kv_a   = (const float*)d_in[5];
  const float* w_k_nope = (const float*)d_in[6];
  const float* w_k_rope = (const float*)d_in[7];
  const float* w_v      = (const float*)d_in[8];
  const float* w_o      = (const float*)d_in[9];

  const size_t M = (size_t)kB * kS;  // 4096

  char* ws = (char*)d_ws;
  auto take = [&](size_t elems) { void* p = ws; ws += elems * 2; return p; };

  void* hid_bf   = take(M * 2048);
  void* wt_q_a   = take(1536 * 2048);   // [1536][2048]
  void* wt_qn    = take(2048 * 1536);   // [2048][1536]
  void* wt_qr    = take(1024 * 1536);
  void* wt_kv_a  = take(512 * 2048);
  void* wt_kn    = take(2048 * 512);
  void* wt_kr    = take(1024 * 512);
  void* wt_v     = take(2048 * 512);
  void* wt_o     = take(2048 * 2048);
  void* q_c      = take(M * 1536);
  void* kv_c     = take(M * 512);
  void* q_nope   = take(M * 2048);
  void* q_rope   = take(M * 1024);
  void* k_nope   = take(M * 2048);
  void* k_rope   = take(M * 1024);
  void* vbuf     = take(M * 2048);
  void* vtb      = take(M * 2048);      // [B][NH][DV][S]
  void* ctx      = take(M * 2048);

  const dim3 blk(256);

  // ---- pre-pack: hidden -> bf16; weights -> bf16 transposed [N][K] ----
  {
    const int n4 = (int)(M * 2048 / 4);
    cvt_bf16<<<(n4 + 255) / 256, blk, 0, stream>>>(hidden, hid_bf, n4);
  }
  auto tw = [&](const float* w, void* wt, int K, int N) {
    transpose_w<<<dim3(N / 32, K / 32), blk, 0, stream>>>(w, wt, K, N);
  };
  tw(w_q_a,    wt_q_a,  2048, 1536);
  tw(w_q_nope, wt_qn,   1536, 2048);
  tw(w_q_rope, wt_qr,   1536, 1024);
  tw(w_kv_a,   wt_kv_a, 2048, 512);
  tw(w_k_nope, wt_kn,   512,  2048);
  tw(w_k_rope, wt_kr,   512,  1024);
  tw(w_v,      wt_v,    512,  2048);
  tw(w_o,      wt_o,    2048, 2048);

  auto grid = [&](int N) { return dim3((unsigned)(M / 128), (unsigned)(N / 128)); };

  // ---- projection GEMMs (all NT, bf16 in, bf16 out) ----
  gemm_nt<1><<<grid(1536), blk, 0, stream>>>(hid_bf, wt_q_a,  q_c,    1536, 2048);
  gemm_nt<1><<<grid(512),  blk, 0, stream>>>(hid_bf, wt_kv_a, kv_c,   512,  2048);
  gemm_nt<1><<<grid(2048), blk, 0, stream>>>(q_c,    wt_qn,   q_nope, 2048, 1536);
  gemm_nt<1><<<grid(1024), blk, 0, stream>>>(q_c,    wt_qr,   q_rope, 1024, 1536);
  gemm_nt<1><<<grid(2048), blk, 0, stream>>>(kv_c,   wt_kn,   k_nope, 2048, 512);
  gemm_nt<1><<<grid(1024), blk, 0, stream>>>(kv_c,   wt_kr,   k_rope, 1024, 512);
  gemm_nt<1><<<grid(2048), blk, 0, stream>>>(kv_c,   wt_v,    vbuf,   2048, 512);

  // ---- RoPE + V transpose ----
  const int pairs = (int)(M * kNH * 32);
  rope_kernel<<<(pairs + 255) / 256, blk, 0, stream>>>(q_rope, pairs);
  rope_kernel<<<(pairs + 255) / 256, blk, 0, stream>>>(k_rope, pairs);
  transpose_v<<<dim3(kB * kNH, kS / 32, kDV / 32), blk, 0, stream>>>(vbuf, vtb);

  // ---- causal flash attention ----
  mla_attn<<<dim3(kB * kNH, kS / 128), blk, 0, stream>>>(
      q_nope, q_rope, k_nope, k_rope, vtb, ctx);

  // ---- output projection ----
  gemm_nt<0><<<grid(2048), blk, 0, stream>>>(ctx, wt_o, d_out, 2048, 2048);
}
